// MANSF_1013612282184
// MI455X (gfx1250) — matrix operand order repaired and checked by reading, not compiled, not run on hardware
//
#include <hip/hip_runtime.h>
#include <hip/hip_bf16.h>
#include <math.h>

#define SS   500
#define DDD  5
#define TT   30
#define FTD  512
#define HH   64
#define G3   192       // 3*H
#define NRELC 5
#define HEADSC 8
#define NCC  2
#define NE   20000

typedef __bf16 bf16_t;
typedef bf16_t v16bf __attribute__((ext_vector_type(16)));
typedef float  v8f   __attribute__((ext_vector_type(8)));

// ---------------------------------------------------------------- utilities
__device__ __forceinline__ float sigm(float x) { return 1.f / (1.f + expf(-x)); }

__device__ __forceinline__ void atomicMaxFloat(float* addr, float v) {
    // order-preserving int encoding trick (valid for mixed signs)
    if (v >= 0.f) atomicMax((int*)addr, __float_as_int(v));
    else          atomicMin((unsigned int*)addr, (unsigned int)__float_as_int(v));
}

// load 8 consecutive floats (32B-aligned) -> bf16 lanes [base..base+7], scaled by msk
__device__ __forceinline__ void load8_bf16(const float* __restrict__ p, float msk,
                                           v16bf& a, int base)
{
    float4 f0 = *(const float4*)(p);
    float4 f1 = *(const float4*)(p + 4);
    a[base + 0] = (bf16_t)(f0.x * msk); a[base + 1] = (bf16_t)(f0.y * msk);
    a[base + 2] = (bf16_t)(f0.z * msk); a[base + 3] = (bf16_t)(f0.w * msk);
    a[base + 4] = (bf16_t)(f1.x * msk); a[base + 5] = (bf16_t)(f1.y * msk);
    a[base + 6] = (bf16_t)(f1.z * msk); a[base + 7] = (bf16_t)(f1.w * msk);
}

__global__ void fill_kernel(float* p, float v, int n) {
    int i = blockIdx.x * blockDim.x + threadIdx.x;
    if (i < n) p[i] = v;
}

// ============================================================== Kernel 1:
// Gi[s] = X[s](150x512) @ tg_Wih[s]^T(512x192) + tg_bih   via bf16 WMMA.
// One wave -> one 16x16 tile of D. K loop of 32. All loads unconditional b128.
__global__ void __launch_bounds__(256) text_ih_wmma(
    const float* __restrict__ X, const float* __restrict__ W,
    const float* __restrict__ bih, float* __restrict__ Gi)
{
    const int MT = 10, NT = 12;                       // 160/16 x 192/16 tiles
    int wid  = blockIdx.x * (blockDim.x >> 5) + (threadIdx.x >> 5);
    int lane = threadIdx.x & 31;
    int s    = wid / (MT * NT);
    if (s >= SS) return;
    int rem  = wid % (MT * NT);
    int mt = rem / NT, nt = rem % NT;
    int half = lane >> 4, lr = lane & 15;

    int m = mt * 16 + lr;
    bool mv = m < (DDD * TT);
    int mc = mv ? m : (DDD * TT - 1);     // clamp: loads always valid
    float msk = mv ? 1.f : 0.f;           // zero contribution post-load
    int n = nt * 16 + lr;

    const float* Arow = X + ((size_t)s * (DDD * TT) + mc) * FTD;
    const float* Brow = W + ((size_t)s * G3 + n) * FTD;   // B[k][n] = W[n][k]

    v8f acc = {0.f, 0.f, 0.f, 0.f, 0.f, 0.f, 0.f, 0.f};
    for (int k0 = 0; k0 < FTD; k0 += 32) {
        v16bf a, b;
        // A 16x32 bf16: lane-half picks K chunks {0..7,16..23} / {8..15,24..31}
        load8_bf16(Arow + k0 + half * 8,      msk, a, 0);
        load8_bf16(Arow + k0 + 16 + half * 8, msk, a, 8);
        // B 32x16 bf16: lane-half picks K chunks {0..7,16..23} / {8..15,24..31}; contiguous in W row n
        load8_bf16(Brow + k0 + half * 8,      1.f, b, 0);
        load8_bf16(Brow + k0 + 16 + half * 8, 1.f, b, 8);
        acc = __builtin_amdgcn_wmma_f32_16x16x32_bf16(false, a, false, b,
                                                      (short)0, acc, false, false);
    }
    int col = nt * 16 + lr;                 // C/D: lane = column
    float bcol = bih[(size_t)s * G3 + col];
    float* Grow = Gi + ((size_t)s * (DDD * TT) + (size_t)mt * 16 + half * 8) * G3 + col;
    if (mt < MT - 1) {
        // full tile: unguarded stores
#pragma unroll
        for (int r = 0; r < 8; ++r) Grow[(size_t)r * G3] = acc[r] + bcol;
    } else {
        // edge tile (rows 144..159, valid < 150)
#pragma unroll
        for (int r = 0; r < 8; ++r) {
            int row = mt * 16 + half * 8 + r;
            if (row < DDD * TT) Grow[(size_t)r * G3] = acc[r] + bcol;
        }
    }
}

// ============================================================== attention tail
// s_t = tanh(last@W1 + b1 + full_t@W2 + b2); score=s@V+bV; softmax; weighted sum.
__device__ void attn_tail(const float* W1, const float* b1, const float* W2,
                          const float* b2, const float* V, float bV,
                          const float* sFull, const float* hlast, int Tm,
                          float* outv, float* q, float* sc, float* red, int tid)
{
    if (tid < HH) {
        float acc = b1[tid] + b2[tid];
        for (int i = 0; i < HH; ++i) acc += hlast[i] * W1[i * HH + tid];
        q[tid] = acc;
    }
    __syncthreads();
    for (int t = 0; t < Tm; ++t) {
        if (tid < HH) {
            float acc = q[tid];
            for (int i = 0; i < HH; ++i) acc += sFull[t * HH + i] * W2[i * HH + tid];
            red[tid] = tanhf(acc) * V[tid];
        }
        __syncthreads();
        for (int off = 32; off > 0; off >>= 1) {
            if (tid < off) red[tid] += red[tid + off];
            __syncthreads();
        }
        if (tid == 0) sc[t] = red[0] + bV;
        __syncthreads();
    }
    if (tid == 0) {
        float mx = sc[0];
        for (int t = 1; t < Tm; ++t) mx = sc[t] > mx ? sc[t] : mx;
        float sum = 0.f;
        for (int t = 0; t < Tm; ++t) { sc[t] = expf(sc[t] - mx); sum += sc[t]; }
        float inv = 1.f / sum;
        for (int t = 0; t < Tm; ++t) sc[t] *= inv;
    }
    __syncthreads();
    if (tid < HH) {
        float acc = 0.f;
        for (int t = 0; t < Tm; ++t) acc += sc[t] * sFull[t * HH + tid];
        outv[tid] = acc;
    }
}

// ============================================================== Kernel 2:
// text GRU scan (Gi precomputed) + attention -> news[s][d][64].
__global__ void __launch_bounds__(192) text_gru_attn(
    const float* __restrict__ Gi, const float* __restrict__ Whh,
    const float* __restrict__ bhh,
    const float* W1, const float* b1, const float* W2, const float* b2,
    const float* V, const float* bV, float* news)
{
    __shared__ float sWhh[HH * G3];     // [j][g] transposed: conflict-free
    __shared__ float sFull[TT * HH];
    __shared__ float sh[HH];
    __shared__ float sgi[G3], sgh[G3];
    __shared__ float q[HH], sc[TT], red[HH];
    int tid = threadIdx.x;
    int s = blockIdx.x / DDD, d = blockIdx.x % DDD;
    const float* Wg = Whh + (size_t)s * G3 * HH;
    for (int i = tid; i < G3 * HH; i += 192) {
        int g = i / HH, j = i % HH;
        sWhh[j * G3 + g] = Wg[i];
    }
    if (tid < HH) sh[tid] = 0.f;
    __syncthreads();
    const float* bh = bhh + (size_t)s * G3;
    for (int t = 0; t < TT; ++t) {
        float acc = bh[tid];
        for (int j = 0; j < HH; ++j) acc += sWhh[j * G3 + tid] * sh[j];
        sgh[tid] = acc;
        sgi[tid] = Gi[((size_t)s * (DDD * TT) + d * TT + t) * G3 + tid];
        __syncthreads();
        if (tid < HH) {
            float r = sigm(sgi[tid] + sgh[tid]);
            float z = sigm(sgi[HH + tid] + sgh[HH + tid]);
            float nn = tanhf(sgi[2 * HH + tid] + r * sgh[2 * HH + tid]);
            float hn = (1.f - z) * nn + z * sh[tid];
            sh[tid] = hn; sFull[t * HH + tid] = hn;
        }
        __syncthreads();
    }
    attn_tail(W1 + (size_t)s * HH * HH, b1 + (size_t)s * HH,
              W2 + (size_t)s * HH * HH, b2 + (size_t)s * HH,
              V + (size_t)s * HH, bV[s], sFull, sh, TT,
              news + ((size_t)s * DDD + d) * HH, q, sc, red, tid);
}

// ============================================================== Kernel 3:
// price GRU (input dim 3, Tm=5) + attention -> xp[s][64]
__global__ void __launch_bounds__(192) price_gru_attn(
    const float* __restrict__ price, const float* __restrict__ Wih,
    const float* __restrict__ bih, const float* __restrict__ Whh,
    const float* __restrict__ bhh,
    const float* W1, const float* b1, const float* W2, const float* b2,
    const float* V, const float* bV, float* xp)
{
    __shared__ float sWhh[HH * G3];
    __shared__ float sFull[DDD * HH];
    __shared__ float sh[HH], sgi[G3], sgh[G3], sx[DDD * 3];
    __shared__ float q[HH], sc[DDD], red[HH];
    int tid = threadIdx.x, s = blockIdx.x;
    const float* Wg = Whh + (size_t)s * G3 * HH;
    for (int i = tid; i < G3 * HH; i += 192) {
        int g = i / HH, j = i % HH;
        sWhh[j * G3 + g] = Wg[i];
    }
    if (tid < DDD * 3) sx[tid] = price[(size_t)s * DDD * 3 + tid];
    if (tid < HH) sh[tid] = 0.f;
    __syncthreads();
    const float* Wi = Wih + (size_t)s * G3 * 3;
    const float* bi = bih + (size_t)s * G3;
    const float* bh = bhh + (size_t)s * G3;
    for (int t = 0; t < DDD; ++t) {
        float gi = bi[tid] + Wi[tid * 3] * sx[t * 3] + Wi[tid * 3 + 1] * sx[t * 3 + 1]
                 + Wi[tid * 3 + 2] * sx[t * 3 + 2];
        float gh = bh[tid];
        for (int j = 0; j < HH; ++j) gh += sWhh[j * G3 + tid] * sh[j];
        sgi[tid] = gi; sgh[tid] = gh;
        __syncthreads();
        if (tid < HH) {
            float r = sigm(sgi[tid] + sgh[tid]);
            float z = sigm(sgi[HH + tid] + sgh[HH + tid]);
            float nn = tanhf(sgi[2 * HH + tid] + r * sgh[2 * HH + tid]);
            float hn = (1.f - z) * nn + z * sh[tid];
            sh[tid] = hn; sFull[t * HH + tid] = hn;
        }
        __syncthreads();
    }
    attn_tail(W1 + (size_t)s * HH * HH, b1 + (size_t)s * HH,
              W2 + (size_t)s * HH * HH, b2 + (size_t)s * HH,
              V + (size_t)s * HH, bV[s], sFull, sh, DDD,
              xp + (size_t)s * HH, q, sc, red, tid);
}

// ============================================================== Kernel 4:
// news GRU (input dim 64, Tm=5) + attention -> tv[s][64]
__global__ void __launch_bounds__(192) news_gru_attn(
    const float* __restrict__ news, const float* __restrict__ Wih,
    const float* __restrict__ bih, const float* __restrict__ Whh,
    const float* __restrict__ bhh,
    const float* W1, const float* b1, const float* W2, const float* b2,
    const float* V, const float* bV, float* tv)
{
    __shared__ float sWhh[HH * G3];
    __shared__ float sFull[DDD * HH];
    __shared__ float sh[HH], sgi[G3], sgh[G3], sn[DDD * HH];
    __shared__ float q[HH], sc[DDD], red[HH];
    int tid = threadIdx.x, s = blockIdx.x;
    const float* Wg = Whh + (size_t)s * G3 * HH;
    for (int i = tid; i < G3 * HH; i += 192) {
        int g = i / HH, j = i % HH;
        sWhh[j * G3 + g] = Wg[i];
    }
    for (int i = tid; i < DDD * HH; i += 192) sn[i] = news[(size_t)s * DDD * HH + i];
    if (tid < HH) sh[tid] = 0.f;
    __syncthreads();
    const float* Wi = Wih + (size_t)s * G3 * HH;
    const float* bi = bih + (size_t)s * G3;
    const float* bh = bhh + (size_t)s * G3;
    for (int t = 0; t < DDD; ++t) {
        float gi = bi[tid];
        for (int j = 0; j < HH; ++j) gi += Wi[tid * HH + j] * sn[t * HH + j];
        float gh = bh[tid];
        for (int j = 0; j < HH; ++j) gh += sWhh[j * G3 + tid] * sh[j];
        sgi[tid] = gi; sgh[tid] = gh;
        __syncthreads();
        if (tid < HH) {
            float r = sigm(sgi[tid] + sgh[tid]);
            float z = sigm(sgi[HH + tid] + sgh[HH + tid]);
            float nn = tanhf(sgi[2 * HH + tid] + r * sgh[2 * HH + tid]);
            float hn = (1.f - z) * nn + z * sh[tid];
            sh[tid] = hn; sFull[t * HH + tid] = hn;
        }
        __syncthreads();
    }
    attn_tail(W1 + (size_t)s * HH * HH, b1 + (size_t)s * HH,
              W2 + (size_t)s * HH * HH, b2 + (size_t)s * HH,
              V + (size_t)s * HH, bV[s], sFull, sh, DDD,
              tv + (size_t)s * HH, q, sc, red, tid);
}

// ============================================================== Kernel 5:
// ft[s][k] = tanh( tv . bl_W[s,k] . xp + bl_b[s,k] ).  524MB stream: wave-per-k,
// float4 loads + prefetch, shuffle reduction.
__global__ void __launch_bounds__(256) bilinear_tanh(
    const float* __restrict__ blW, const float* __restrict__ blb,
    const float* __restrict__ tv, const float* __restrict__ xp, float* ft)
{
    __shared__ float stv[HH], sxp[HH];
    int tid = threadIdx.x, s = blockIdx.x;
    if (tid < HH) { stv[tid] = tv[(size_t)s * HH + tid]; sxp[tid] = xp[(size_t)s * HH + tid]; }
    __syncthreads();
    int wave = tid >> 5, lane = tid & 31;
    const float* A = blW + (size_t)s * HH * HH * HH;
    for (int k = wave; k < HH; k += 8) {
        const float4* Ak = (const float4*)(A + (size_t)k * HH * HH);
        float acc = 0.f;
        for (int qq = 0; qq < 32; ++qq) {
            int idx = qq * 32 + lane;
            __builtin_prefetch(Ak + idx + 512, 0, 1);
            float4 v = Ak[idx];
            int flat = idx * 4;
            int i = flat >> 6, j = flat & 63;
            acc += stv[i] * (v.x * sxp[j] + v.y * sxp[j + 1] + v.z * sxp[j + 2] + v.w * sxp[j + 3]);
        }
        for (int off = 16; off > 0; off >>= 1) acc += __shfl_xor(acc, off, 32);
        if (lane == 0) ft[(size_t)s * HH + k] = tanhf(acc + blb[(size_t)s * HH + k]);
    }
}

// ============================================================== Kernel 6: out1
__global__ void out1_kernel(const float* ft, const float* lxW, const float* lxb, float* out1) {
    int i = blockIdx.x * blockDim.x + threadIdx.x;
    if (i >= SS * NCC) return;
    int s = i / NCC, c = i % NCC;
    float acc = lxb[c];
    for (int j = 0; j < HH; ++j) acc += ft[s * HH + j] * lxW[j * NCC + c];
    out1[i] = tanhf(acc);
}

// ============================================================== transpose g1_W
// g1W [r][k][n] (k<64, n<512) -> g1Wt [r][n][k]: makes WMMA B-fragment contiguous.
__global__ void transpose_g1w(const float* __restrict__ g1W, float* __restrict__ g1Wt) {
    int idx = blockIdx.x * blockDim.x + threadIdx.x;
    if (idx >= NRELC * HH * HEADSC * HH) return;
    int r = idx / (HH * HEADSC * HH);
    int rem = idx % (HH * HEADSC * HH);
    int k = rem / (HEADSC * HH);
    int n = rem % (HEADSC * HH);
    g1Wt[((size_t)r * (HEADSC * HH) + n) * HH + k] = g1W[idx];
}

// ============================================================== Kernel 7:
// xr1[n][r][:] = ft[n] @ g1_W[r] (64 -> 512), bf16 WMMA with transposed B.
__global__ void __launch_bounds__(256) rgat1_lin_wmma(
    const float* __restrict__ ft, const float* __restrict__ g1Wt, float* __restrict__ xr1)
{
    const int MT = 32, NT = 32;
    int wid  = blockIdx.x * (blockDim.x >> 5) + (threadIdx.x >> 5);
    int lane = threadIdx.x & 31;
    int r = wid / (MT * NT);
    if (r >= NRELC) return;
    int rem = wid % (MT * NT);
    int mt = rem / NT, nt = rem % NT;
    int half = lane >> 4, lr = lane & 15;
    int m = mt * 16 + lr;
    bool mv = m < SS;
    int mc = mv ? m : SS - 1;
    float msk = mv ? 1.f : 0.f;
    int n = nt * 16 + lr;
    const float* Arow = ft + (size_t)mc * HH;
    const float* Brow = g1Wt + ((size_t)r * (HEADSC * HH) + n) * HH;  // B[k][n] contiguous in k
    v8f acc = {0.f, 0.f, 0.f, 0.f, 0.f, 0.f, 0.f, 0.f};
    for (int k0 = 0; k0 < HH; k0 += 32) {
        v16bf a, b;
        load8_bf16(Arow + k0 + half * 8,      msk, a, 0);
        load8_bf16(Arow + k0 + 16 + half * 8, msk, a, 8);
        load8_bf16(Brow + k0 + half * 8,      1.f, b, 0);
        load8_bf16(Brow + k0 + 16 + half * 8, 1.f, b, 8);
        acc = __builtin_amdgcn_wmma_f32_16x16x32_bf16(false, a, false, b,
                                                      (short)0, acc, false, false);
    }
    int col = nt * 16 + lr;
    float* Orow = xr1 + (((size_t)mt * 16 + half * 8) * NRELC + r) * (HEADSC * HH) + col;
    const size_t ostride = (size_t)NRELC * HEADSC * HH;
    if (mt < MT - 1) {
        // full tile (rows < 496): unguarded stores
#pragma unroll
        for (int rr = 0; rr < 8; ++rr) Orow[(size_t)rr * ostride] = acc[rr];
    } else {
        // edge tile (rows 496..511, valid < 500)
#pragma unroll
        for (int rr = 0; rr < 8; ++rr) {
            int row = mt * 16 + half * 8 + rr;
            if (row < SS) Orow[(size_t)rr * ostride] = acc[rr];
        }
    }
}

// ============================================================== RGAT layer 1 edges
__global__ void edge1_score(const int* __restrict__ ei, const int* __restrict__ et,
                            const float* __restrict__ xr1, const float* __restrict__ g1a,
                            float* esc, float* m1)
{
    int idx = blockIdx.x * blockDim.x + threadIdx.x;
    if (idx >= NE * HEADSC) return;
    int e = idx / HEADSC, h = idx % HEADSC;
    int srcn = ei[e], dstn = ei[NE + e], r = et[e];
    const float* a  = g1a + ((size_t)r * HEADSC + h) * (2 * HH);
    const float* hs = xr1 + ((size_t)srcn * NRELC + r) * (HEADSC * HH) + h * HH;
    const float* hd = xr1 + ((size_t)dstn * NRELC + r) * (HEADSC * HH) + h * HH;
    float acc = 0.f;
    for (int o = 0; o < HH; ++o) acc += a[o] * hd[o] + a[HH + o] * hs[o];
    if (acc < 0.f) acc *= 0.2f;
    esc[idx] = acc;
    atomicMaxFloat(&m1[dstn * HEADSC + h], acc);
}

__global__ void edge1_accum(const int* __restrict__ ei, const int* __restrict__ et,
                            const float* __restrict__ xr1, const float* __restrict__ esc,
                            const float* __restrict__ m1, float* den1, float* msg1)
{
    int idx = blockIdx.x * blockDim.x + threadIdx.x;
    if (idx >= NE * HEADSC) return;
    int e = idx / HEADSC, h = idx % HEADSC;
    int srcn = ei[e], dstn = ei[NE + e], r = et[e];
    float ee = expf(esc[idx] - m1[dstn * HEADSC + h]);
    atomicAdd(&den1[dstn * HEADSC + h], ee);
    const float* hs = xr1 + ((size_t)srcn * NRELC + r) * (HEADSC * HH) + h * HH;
    float* mo = msg1 + (size_t)dstn * (HEADSC * HH) + h * HH;
    for (int o = 0; o < HH; ++o) atomicAdd(&mo[o], ee * hs[o]);
}

__global__ void edge1_norm(const float* msg1, const float* den1, float* x1) {
    int i = blockIdx.x * blockDim.x + threadIdx.x;
    if (i >= SS * HEADSC * HH) return;
    int n = i / (HEADSC * HH), c = i % (HEADSC * HH);
    x1[i] = msg1[i] / (den1[n * HEADSC + c / HH] + 1e-9f);
}

// ============================================================== RGAT layer 2
__global__ void rgat2_lin(const float* __restrict__ x1, const float* __restrict__ g2W,
                          float* xr2)
{
    int idx = blockIdx.x * blockDim.x + threadIdx.x;
    if (idx >= SS * NRELC * NCC) return;
    int c = idx % NCC, r = (idx / NCC) % NRELC, n = idx / (NCC * NRELC);
    const float* w = g2W + (size_t)r * (HEADSC * HH) * NCC;
    float acc = 0.f;
    for (int k = 0; k < HEADSC * HH; ++k) acc += x1[(size_t)n * (HEADSC * HH) + k] * w[k * NCC + c];
    xr2[idx] = acc;
}

__global__ void edge2_score(const int* __restrict__ ei, const int* __restrict__ et,
                            const float* __restrict__ xr2, const float* __restrict__ g2a,
                            float* e2, float* m2)
{
    int e = blockIdx.x * blockDim.x + threadIdx.x;
    if (e >= NE) return;
    int srcn = ei[e], dstn = ei[NE + e], r = et[e];
    const float* a  = g2a + (size_t)r * 2 * NCC;
    const float* hs = xr2 + ((size_t)srcn * NRELC + r) * NCC;
    const float* hd = xr2 + ((size_t)dstn * NRELC + r) * NCC;
    float acc = 0.f;
    for (int o = 0; o < NCC; ++o) acc += a[o] * hd[o] + a[NCC + o] * hs[o];
    if (acc < 0.f) acc *= 0.2f;
    e2[e] = acc;
    atomicMaxFloat(&m2[dstn], acc);
}

__global__ void edge2_accum(const int* __restrict__ ei, const int* __restrict__ et,
                            const float* __restrict__ xr2, const float* __restrict__ e2,
                            const float* __restrict__ m2, float* den2, float* msg2)
{
    int e = blockIdx.x * blockDim.x + threadIdx.x;
    if (e >= NE) return;
    int srcn = ei[e], dstn = ei[NE + e], r = et[e];
    float ee = expf(e2[e] - m2[dstn]);
    atomicAdd(&den2[dstn], ee);
    const float* hs = xr2 + ((size_t)srcn * NRELC + r) * NCC;
    for (int o = 0; o < NCC; ++o) atomicAdd(&msg2[dstn * NCC + o], ee * hs[o]);
}

__global__ void final_kernel(const float* msg2, const float* den2, const float* out1,
                             float* out)
{
    int s = blockIdx.x * blockDim.x + threadIdx.x;
    if (s >= SS) return;
    float inv = 1.f / (den2[s] + 1e-9f);
    float g0 = msg2[s * 2] * inv, g1 = msg2[s * 2 + 1] * inv;
    float x0 = (g0 > 0.f ? g0 : expf(g0) - 1.f) + out1[s * 2];
    float x1 = (g1 > 0.f ? g1 : expf(g1) - 1.f) + out1[s * 2 + 1];
    float mx = x0 > x1 ? x0 : x1;
    float e0 = expf(x0 - mx), e1 = expf(x1 - mx);
    float d = 1.f / (e0 + e1);
    out[s * 2] = e0 * d; out[s * 2 + 1] = e1 * d;
}

// ============================================================== launcher
extern "C" void kernel_launch(void* const* d_in, const int* in_sizes, int n_in,
                              void* d_out, int out_size, void* d_ws, size_t ws_size,
                              hipStream_t stream)
{
    const float* text  = (const float*)d_in[0];
    const float* price = (const float*)d_in[1];
    const int*   ei    = (const int*)d_in[2];
    const int*   et    = (const int*)d_in[3];
    const float* gpWih = (const float*)d_in[4];
    const float* gpWhh = (const float*)d_in[5];
    const float* gpbih = (const float*)d_in[6];
    const float* gpbhh = (const float*)d_in[7];
    const float* tgWih = (const float*)d_in[8];
    const float* tgWhh = (const float*)d_in[9];
    const float* tgbih = (const float*)d_in[10];
    const float* tgbhh = (const float*)d_in[11];
    const float* gtWih = (const float*)d_in[12];
    const float* gtWhh = (const float*)d_in[13];
    const float* gtbih = (const float*)d_in[14];
    const float* gtbhh = (const float*)d_in[15];
    const float* apW1 = (const float*)d_in[16]; const float* apb1 = (const float*)d_in[17];
    const float* apW2 = (const float*)d_in[18]; const float* apb2 = (const float*)d_in[19];
    const float* apV  = (const float*)d_in[20]; const float* apbV = (const float*)d_in[21];
    const float* atW1 = (const float*)d_in[22]; const float* atb1 = (const float*)d_in[23];
    const float* atW2 = (const float*)d_in[24]; const float* atb2 = (const float*)d_in[25];
    const float* atV  = (const float*)d_in[26]; const float* atbV = (const float*)d_in[27];
    const float* anW1 = (const float*)d_in[28]; const float* anb1 = (const float*)d_in[29];
    const float* anW2 = (const float*)d_in[30]; const float* anb2 = (const float*)d_in[31];
    const float* anV  = (const float*)d_in[32]; const float* anbV = (const float*)d_in[33];
    const float* blW  = (const float*)d_in[34]; const float* blb  = (const float*)d_in[35];
    const float* lxW  = (const float*)d_in[36]; const float* lxb  = (const float*)d_in[37];
    const float* g1W  = (const float*)d_in[38]; const float* g1a  = (const float*)d_in[39];
    const float* g2W  = (const float*)d_in[40]; const float* g2a  = (const float*)d_in[41];
    float* out = (float*)d_out;

    // workspace layout (floats)
    float* ws = (float*)d_ws;
    float* Gi   = ws;                           // 500*150*192 = 14,400,000
    float* news = Gi   + (size_t)SS * DDD * TT * G3;
    float* xp   = news + (size_t)SS * DDD * HH; // 160,000
    float* tv   = xp   + (size_t)SS * HH;
    float* ft   = tv   + (size_t)SS * HH;
    float* out1 = ft   + (size_t)SS * HH;
    float* xr1  = out1 + (size_t)SS * NCC;      // 500*5*512
    float* esc  = xr1  + (size_t)SS * NRELC * HEADSC * HH;  // 160,000
    float* m1   = esc  + (size_t)NE * HEADSC;
    float* den1 = m1   + (size_t)SS * HEADSC;
    float* msg1 = den1 + (size_t)SS * HEADSC;
    float* x1   = msg1 + (size_t)SS * HEADSC * HH;
    float* xr2  = x1   + (size_t)SS * HEADSC * HH;
    float* e2   = xr2  + (size_t)SS * NRELC * NCC;
    float* m2   = e2   + (size_t)NE;
    float* den2 = m2   + (size_t)SS;
    float* msg2 = den2 + (size_t)SS;
    float* g1Wt = msg2 + (size_t)SS * NCC;      // 5*512*64 = 163,840

    // ---- phase 1: per-stock pipelines
    text_ih_wmma<<<7500, 256, 0, stream>>>(text, tgWih, tgbih, Gi);
    price_gru_attn<<<SS, 192, 0, stream>>>(price, gpWih, gpbih, gpWhh, gpbhh,
                                           apW1, apb1, apW2, apb2, apV, apbV, xp);
    text_gru_attn<<<SS * DDD, 192, 0, stream>>>(Gi, tgWhh, tgbhh,
                                                atW1, atb1, atW2, atb2, atV, atbV, news);
    news_gru_attn<<<SS, 192, 0, stream>>>(news, gtWih, gtbih, gtWhh, gtbhh,
                                          anW1, anb1, anW2, anb2, anV, anbV, tv);
    bilinear_tanh<<<SS, 256, 0, stream>>>(blW, blb, tv, xp, ft);
    out1_kernel<<<(SS * NCC + 255) / 256, 256, 0, stream>>>(ft, lxW, lxb, out1);

    // ---- phase 2: RGAT
    fill_kernel<<<(SS * HEADSC + 255) / 256, 256, 0, stream>>>(m1, -1e30f, SS * HEADSC);
    fill_kernel<<<(SS * HEADSC + 255) / 256, 256, 0, stream>>>(den1, 0.f, SS * HEADSC);
    fill_kernel<<<(SS * HEADSC * HH + 255) / 256, 256, 0, stream>>>(msg1, 0.f, SS * HEADSC * HH);
    fill_kernel<<<(SS + 255) / 256, 256, 0, stream>>>(m2, -1e30f, SS);
    fill_kernel<<<(SS + 255) / 256, 256, 0, stream>>>(den2, 0.f, SS);
    fill_kernel<<<(SS * NCC + 255) / 256, 256, 0, stream>>>(msg2, 0.f, SS * NCC);

    transpose_g1w<<<(NRELC * HH * HEADSC * HH + 255) / 256, 256, 0, stream>>>(g1W, g1Wt);
    rgat1_lin_wmma<<<640, 256, 0, stream>>>(ft, g1Wt, xr1);
    edge1_score<<<(NE * HEADSC + 255) / 256, 256, 0, stream>>>(ei, et, xr1, g1a, esc, m1);
    edge1_accum<<<(NE * HEADSC + 255) / 256, 256, 0, stream>>>(ei, et, xr1, esc, m1, den1, msg1);
    edge1_norm<<<(SS * HEADSC * HH + 255) / 256, 256, 0, stream>>>(msg1, den1, x1);

    rgat2_lin<<<(SS * NRELC * NCC + 255) / 256, 256, 0, stream>>>(x1, g2W, xr2);
    edge2_score<<<(NE + 255) / 256, 256, 0, stream>>>(ei, et, xr2, g2a, e2, m2);
    edge2_accum<<<(NE + 255) / 256, 256, 0, stream>>>(ei, et, xr2, e2, m2, den2, msg2);
    final_kernel<<<(SS + 255) / 256, 256, 0, stream>>>(msg2, den2, out1, out);
}
